// PSM_48155173322926
// MI455X (gfx1250) — matrix operand order repaired and checked by reading, not compiled, not run on hardware
//
#include <hip/hip_runtime.h>

typedef __attribute__((ext_vector_type(16))) _Float16 v16h;
typedef __attribute__((ext_vector_type(8)))  float    v8f;

#define N_NODES 100000
#define N_EDGES 1600000
#define N_EL    (N_EDGES + N_NODES)
#define HID     64
#define DIN     512

static __device__ __forceinline__ float leaky02(float v) {
    return v > 0.0f ? v : 0.2f * v;
}

static __device__ __forceinline__ void atomicMaxF(float* addr, float val) {
    int* ai = (int*)addr;
    int cur = __float_as_int(*addr);
    while (__int_as_float(cur) < val) {
        int prev = atomicCAS(ai, cur, __float_as_int(val));
        if (prev == cur) break;
        cur = prev;
    }
}

// ---------------------------------------------------------------------------
// Weight swizzle: W[k x m] f32 (row-major) -> f16 fragments in WMMA B-lane
// order. Fragment element j of lane l for K-block kb, 16-col tile ct holds
// W[kb*32 + (l>>4)*8 + j + (j>=8 ? 8 : 0)][ct*16 + (l&15)].
// Flat layout: Ws[((kb*ncol + ct)*32 + l)*16 + j]  (each lane's 16 f16 are
// contiguous -> two global_load_b128 per fragment in the GEMM).
// ---------------------------------------------------------------------------
__global__ void swizzle_w_kernel(const float* __restrict__ W, _Float16* __restrict__ Ws,
                                 int k, int m)
{
    const int ncol = m >> 4;
    const int total = (k >> 5) * ncol * 32;
    int t = blockIdx.x * blockDim.x + threadIdx.x;
    if (t >= total) return;
    const int l  = t & 31;
    const int ct = (t >> 5) % ncol;
    const int kb = (t >> 5) / ncol;
    const int kbase = kb * 32 + (l >> 4) * 8;
    const int col   = ct * 16 + (l & 15);
    _Float16* dst = Ws + (size_t)t * 16;
#pragma unroll
    for (int j = 0; j < 16; ++j)
        dst[j] = (_Float16)W[(size_t)(kbase + j + (j >= 8 ? 8 : 0)) * m + col];
}

// ---------------------------------------------------------------------------
// WMMA GEMM: C[n x m] = A[n x k] @ W[k x m] (+ bias[m]).
// One wave computes a 16-row x 64-col slab: A fragment loaded/converted once
// per K-step, 4 WMMAs into 4 accumulators (4x A reuse).
// Requires n%16==0, k%32==0, m%64==0. W pre-swizzled to f16 (above).
// A fragment (16-bit 16x32): lane l: row=l&15, half=l>>4;
//   elem j<8 -> K=k0+half*8+j ; elem j>=8 -> K=k0+16+half*8+(j-8)
// C (f32 16x16): vgpr r: lanes0-15 M=r, lanes16-31 M=8+r; N=lane&15.
// ---------------------------------------------------------------------------
__global__ __launch_bounds__(256)
void gemm_wmma4_kernel(const float* __restrict__ A, const _Float16* __restrict__ Wsw,
                       const float* __restrict__ bias, float* __restrict__ C,
                       int n, int k, int m)
{
    const int lane = threadIdx.x & 31;
    const int wid  = blockIdx.x * 8 + (threadIdx.x >> 5);
    const int ncol = m >> 4;                 // 16-col tiles
    const int ngrp = ncol >> 2;              // 64-col groups
    if (wid >= (n >> 4) * ngrp) return;

    const int rowBase = (wid / ngrp) << 4;
    const int grp     = wid % ngrp;
    const int half = lane >> 4;
    const int l16  = lane & 15;
    const int arow = rowBase + l16;

    v8f acc[4] = {};
    for (int k0 = 0; k0 < k; k0 += 32) {
        const float* ap = A + (size_t)arow * k + k0 + half * 8;
        v16h a;
#pragma unroll
        for (int j = 0; j < 8; ++j) a[j] = (_Float16)ap[j];
#pragma unroll
        for (int j = 0; j < 8; ++j) a[8 + j] = (_Float16)ap[16 + j];

        const _Float16* bp = Wsw +
            ((size_t)(((k0 >> 5) * ncol) + (grp << 2)) * 32 + lane) * 16;
#pragma unroll
        for (int t = 0; t < 4; ++t) {
            v16h b = *(const v16h*)(bp + (size_t)t * 32 * 16);
            acc[t] = __builtin_amdgcn_wmma_f32_16x16x32_f16(
                false, a, false, b, (short)0, acc[t], false, false);
        }
    }

#pragma unroll
    for (int t = 0; t < 4; ++t) {
        const int col = ((grp << 2) + t) * 16 + l16;
        const float bv = bias ? bias[col] : 0.0f;
        float* cp = C + (size_t)(rowBase + half * 8) * m + col;
#pragma unroll
        for (int r = 0; r < 8; ++r) cp[(size_t)r * m] = acc[t][r] + bv;
    }
}

// ---------------------------------------------------------------------------
__global__ void fill_f32_kernel(float* __restrict__ p, float v, long long n) {
    long long i = (long long)blockIdx.x * blockDim.x + threadIdx.x;
    if (i < n) p[i] = v;
}

__global__ void alpha_kernel(const float* __restrict__ hw,
                             const float* __restrict__ a_s, const float* __restrict__ a_d,
                             float* __restrict__ alpha_s, float* __restrict__ alpha_d, int n)
{
    int i = blockIdx.x * blockDim.x + threadIdx.x;
    if (i >= n) return;
    const float* h = hw + (size_t)i * HID;
    float s = 0.0f, d = 0.0f;
#pragma unroll 8
    for (int c = 0; c < HID; ++c) { float v = h[c]; s += v * a_s[c]; d += v * a_d[c]; }
    alpha_s[i] = s; alpha_d[i] = d;
}

__global__ void edge_max_kernel(const int* __restrict__ ei,
                                const float* __restrict__ as, const float* __restrict__ ad,
                                float* __restrict__ mseg, int nE, int nN)
{
    int e = blockIdx.x * blockDim.x + threadIdx.x;
    if (e >= nE + nN) return;
    int s, d;
    if (e < nE) { s = ei[e]; d = ei[nE + e]; } else { s = d = e - nE; }
    atomicMaxF(&mseg[d], leaky02(as[s] + ad[d]));
}

__global__ void edge_expsum_kernel(const int* __restrict__ ei,
                                   const float* __restrict__ as, const float* __restrict__ ad,
                                   const float* __restrict__ mseg,
                                   float* __restrict__ wbuf, float* __restrict__ sseg,
                                   int nE, int nN)
{
    int e = blockIdx.x * blockDim.x + threadIdx.x;
    if (e >= nE + nN) return;
    int s, d;
    if (e < nE) { s = ei[e]; d = ei[nE + e]; } else { s = d = e - nE; }
    float w = __expf(leaky02(as[s] + ad[d]) - mseg[d]);
    wbuf[e] = w;
    atomicAdd(&sseg[d], w);
}

__global__ void init_bias_kernel(float* __restrict__ out, const float* __restrict__ b,
                                 long long total, int m)
{
    long long i = (long long)blockIdx.x * blockDim.x + threadIdx.x;
    if (i < total) out[i] = b[(int)(i % m)];
}

// one edge per wave; each lane handles channels (lane, lane+32)
__global__ __launch_bounds__(256)
void edge_aggr_kernel(const int* __restrict__ ei,
                      const float* __restrict__ wbuf, const float* __restrict__ sseg,
                      const float* __restrict__ hw, float* __restrict__ out,
                      int nE, int nN)
{
    long long t = (long long)blockIdx.x * blockDim.x + threadIdx.x;
    long long e = t >> 5;
    if (e >= (long long)(nE + nN)) return;
    int lane = (int)(t & 31);
    int s, d;
    if (e < nE) { s = ei[e]; d = ei[(size_t)nE + e]; } else { s = d = (int)(e - nE); }
    float coef = wbuf[e] / (sseg[d] + 1e-16f);
    const float* hs = hw + (size_t)s * HID;
    float* od = out + (size_t)d * HID;
    atomicAdd(&od[lane],      coef * hs[lane]);
    atomicAdd(&od[lane + 32], coef * hs[lane + 32]);
}

__global__ __launch_bounds__(256)
void bn_stats_kernel(const float* __restrict__ h, float* __restrict__ sum,
                     float* __restrict__ sumsq, long long total, int m)
{
    __shared__ float ls[512];
    __shared__ float lq[512];
    for (int i = threadIdx.x; i < m; i += blockDim.x) { ls[i] = 0.0f; lq[i] = 0.0f; }
    __syncthreads();
    long long stride = (long long)gridDim.x * blockDim.x;
    for (long long i = (long long)blockIdx.x * blockDim.x + threadIdx.x; i < total; i += stride) {
        float v = h[i];
        int c = (int)(i % m);
        atomicAdd(&ls[c], v);
        atomicAdd(&lq[c], v * v);
    }
    __syncthreads();
    for (int i = threadIdx.x; i < m; i += blockDim.x) {
        atomicAdd(&sum[i], ls[i]);
        atomicAdd(&sumsq[i], lq[i]);
    }
}

__global__ void bn_relu_kernel(const float* __restrict__ h,
                               const float* __restrict__ sum, const float* __restrict__ sumsq,
                               const float* __restrict__ g, const float* __restrict__ beta,
                               float* __restrict__ out, long long total, int m, float invN)
{
    long long i = (long long)blockIdx.x * blockDim.x + threadIdx.x;
    if (i >= total) return;
    int c = (int)(i % m);
    float mu  = sum[c] * invN;
    float var = sumsq[c] * invN - mu * mu;
    float v = (h[i] - mu) * rsqrtf(var + 1e-5f) * g[c] + beta[c];
    out[i] = v > 0.0f ? v : 0.0f;
}

__global__ void l2n64_kernel(const float* __restrict__ z, float* __restrict__ zn, int n)
{
    int i = blockIdx.x * blockDim.x + threadIdx.x;
    if (i >= n) return;
    const float* r = z + (size_t)i * HID;
    float s = 0.0f;
#pragma unroll 8
    for (int c = 0; c < HID; ++c) s += r[c] * r[c];
    float inv = 1.0f / fmaxf(sqrtf(s), 1e-12f);
    float* o = zn + (size_t)i * HID;
#pragma unroll 8
    for (int c = 0; c < HID; ++c) o[c] = r[c] * inv;
}

__global__ void cbn_kernel(const float* __restrict__ cb,
                           float* __restrict__ cbn, float* __restrict__ cbnT)
{
    int c = threadIdx.x;
    if (c >= 64) return;
    float s = 0.0f;
    for (int d = 0; d < 64; ++d) { float v = cb[c * 64 + d]; s += v * v; }
    float inv = 1.0f / fmaxf(sqrtf(s), 1e-12f);
    for (int d = 0; d < 64; ++d) {
        float v = cb[c * 64 + d] * inv;
        cbn[c * 64 + d] = v;
        cbnT[d * 64 + c] = v;     // [dim][code] for the score GEMM
    }
}

// argmin of d = 2 - 2*score  ==  first strict argmax of score
__global__ void argmax_kernel(const float* __restrict__ scores, int* __restrict__ idx, int n)
{
    int i = blockIdx.x * blockDim.x + threadIdx.x;
    if (i >= n) return;
    const float* r = scores + (size_t)i * 64;
    float best = r[0]; int bi = 0;
    for (int c = 1; c < 64; ++c) { float v = r[c]; if (v > best) { best = v; bi = c; } }
    idx[i] = bi;
}

__global__ void vq_kernel(const float* __restrict__ zn, const float* __restrict__ cbn,
                          const int* __restrict__ idx, const unsigned char* __restrict__ mask,
                          float* __restrict__ e_out, float* __restrict__ e_mask,
                          int* __restrict__ mi, float* __restrict__ acc, int n)
{
    int i = blockIdx.x * blockDim.x + threadIdx.x;
    if (i >= n) return;
    int id = idx[i];
    int m = mask[id] ? 1 : 0;
    mi[i] = m;
    const float* q = cbn + (size_t)id * 64;
    const float* z = zn + (size_t)i * 64;
    float s = 0.0f;
#pragma unroll 8
    for (int c = 0; c < 64; ++c) {
        float qc = q[c];
        float d = qc - z[c];
        s += d * d;
        e_out[(size_t)i * 64 + c]  = qc;            // e = zn + sg(q - zn) == q
        e_mask[(size_t)i * 64 + c] = m ? 0.0f : qc; // masked decoder input
    }
    atomicAdd(&acc[0], s);
    if (m) atomicAdd(&acc[3], 1.0f);
}

__global__ __launch_bounds__(256)
void mse_kernel(const float* __restrict__ a, const float* __restrict__ b,
                float* __restrict__ acc, long long total)
{
    long long stride = (long long)gridDim.x * blockDim.x;
    float s = 0.0f;
    for (long long i = (long long)blockIdx.x * blockDim.x + threadIdx.x; i < total; i += stride) {
        float d = a[i] - b[i];
        s += d * d;
    }
    for (int o = 16; o > 0; o >>= 1) s += __shfl_down(s, o, 32);
    if ((threadIdx.x & 31) == 0) atomicAdd(&acc[1], s);
}

// one wave per node row (512 cols); masked cosine loss
__global__ __launch_bounds__(256)
void maskloss_kernel(const float* __restrict__ xmr, const float* __restrict__ x,
                     const int* __restrict__ mi, float* __restrict__ acc, int n)
{
    long long t = (long long)blockIdx.x * blockDim.x + threadIdx.x;
    int row = (int)(t >> 5);
    int lane = (int)(t & 31);
    if (row >= n) return;
    if (!mi[row]) return;
    const float* a = xmr + (size_t)row * DIN;
    const float* b = x   + (size_t)row * DIN;
    float dot = 0.0f, na = 0.0f, nb = 0.0f;
    for (int c = lane; c < DIN; c += 32) {
        float u = a[c], v = b[c];
        dot += u * v; na += u * u; nb += v * v;
    }
    for (int o = 16; o > 0; o >>= 1) {
        dot += __shfl_down(dot, o, 32);
        na  += __shfl_down(na, o, 32);
        nb  += __shfl_down(nb, o, 32);
    }
    if (lane == 0) {
        float cs = dot / (fmaxf(sqrtf(na), 1e-12f) * fmaxf(sqrtf(nb), 1e-12f));
        float per = (1.0f - cs) * (1.0f - cs);
        atomicAdd(&acc[2], per);
    }
}

__global__ void finalize_kernel(const float* __restrict__ acc, float* __restrict__ out)
{
    if (threadIdx.x == 0 && blockIdx.x == 0) {
        out[0] = 1.25f * acc[0] / (float)((long long)N_NODES * HID);   // vq_loss, (1+CC)*mse
        out[1] = acc[1] / (float)((long long)N_NODES * DIN);           // recon_loss
        out[2] = acc[2] / (acc[3] + 1e-12f);                           // mask_loss
    }
}

// ---------------------------------------------------------------------------
// Host side
// ---------------------------------------------------------------------------
struct GatP { const float* W; const float* a_s; const float* a_d; const float* b;
              _Float16* Wsw; };
struct FcP  { const float* W; const float* b; const float* g; const float* beta;
              _Float16* Wsw; };

struct Scratch {
    float *hw, *gat_out, *h, *zn, *emask, *scores;
    float *alpha_s, *alpha_d, *mseg, *sseg, *wbuf;
    int *idx, *mi;
    float *cbn, *cbnT, *stats, *acc, *big;
    _Float16 *cbnT_sw;
    const int* ei;
};

static inline int cdiv(long long a, int b) { return (int)((a + b - 1) / b); }

static void run_block(const float* in, int din, int dout,
                      const GatP& g, const FcP& f, float* out,
                      const Scratch& S, hipStream_t st)
{
    // 1. hw = in @ Wg  (WMMA, 16x64 slab per wave)
    {
        int waves = (N_NODES / 16) * (HID / 64);
        gemm_wmma4_kernel<<<(waves + 7) / 8, 256, 0, st>>>(in, g.Wsw, nullptr, S.hw,
                                                           N_NODES, din, HID);
    }
    // 2. per-node attention scalars
    alpha_kernel<<<cdiv(N_NODES, 256), 256, 0, st>>>(S.hw, g.a_s, g.a_d,
                                                     S.alpha_s, S.alpha_d, N_NODES);
    // 3. segment max over dst
    fill_f32_kernel<<<cdiv(N_NODES, 256), 256, 0, st>>>(S.mseg, -3.0e38f, N_NODES);
    edge_max_kernel<<<cdiv(N_EL, 256), 256, 0, st>>>(S.ei, S.alpha_s, S.alpha_d,
                                                     S.mseg, N_EDGES, N_NODES);
    // 4. exp weights + segment sum
    fill_f32_kernel<<<cdiv(N_NODES, 256), 256, 0, st>>>(S.sseg, 0.0f, N_NODES);
    edge_expsum_kernel<<<cdiv(N_EL, 256), 256, 0, st>>>(S.ei, S.alpha_s, S.alpha_d,
                                                        S.mseg, S.wbuf, S.sseg,
                                                        N_EDGES, N_NODES);
    // 5. weighted aggregation into gat_out (pre-seeded with bias)
    init_bias_kernel<<<cdiv((long long)N_NODES * HID, 256), 256, 0, st>>>(
        S.gat_out, g.b, (long long)N_NODES * HID, HID);
    edge_aggr_kernel<<<cdiv((long long)N_EL * 32, 256), 256, 0, st>>>(
        S.ei, S.wbuf, S.sseg, S.hw, S.gat_out, N_EDGES, N_NODES);
    // 6. FC: gat_out @ Wf + bf  (WMMA)
    float* fcbuf = (dout == HID) ? S.hw : S.big;
    {
        int waves = (N_NODES / 16) * (dout / 64);
        gemm_wmma4_kernel<<<(waves + 7) / 8, 256, 0, st>>>(S.gat_out, f.Wsw, f.b, fcbuf,
                                                           N_NODES, HID, dout);
    }
    // 7. BatchNorm (train, biased var) + ReLU
    fill_f32_kernel<<<cdiv(2 * dout, 256), 256, 0, st>>>(S.stats, 0.0f, 2 * dout);
    long long total = (long long)N_NODES * dout;
    bn_stats_kernel<<<1024, 256, 0, st>>>(fcbuf, S.stats, S.stats + dout, total, dout);
    bn_relu_kernel<<<cdiv(total, 256), 256, 0, st>>>(fcbuf, S.stats, S.stats + dout,
                                                     f.g, f.beta, out, total, dout,
                                                     1.0f / (float)N_NODES);
}

extern "C" void kernel_launch(void* const* d_in, const int* in_sizes, int n_in,
                              void* d_out, int out_size, void* d_ws, size_t ws_size,
                              hipStream_t stream)
{
    (void)in_sizes; (void)n_in; (void)out_size; (void)ws_size;

    // setup_inputs order: x, params (keys sorted, leaves sorted), edge_index, mask
    const float* x        = (const float*)d_in[0];
    const float* codebook = (const float*)d_in[1];
    // *_f leaves sorted: W, b, beta, g ; *_g leaves sorted: W, a_d, a_s, b
    FcP  dec1_f{(const float*)d_in[2],  (const float*)d_in[3],  (const float*)d_in[5],  (const float*)d_in[4],  nullptr};
    GatP dec1_g{(const float*)d_in[6],  (const float*)d_in[8],  (const float*)d_in[7],  (const float*)d_in[9],  nullptr};
    FcP  dec2_f{(const float*)d_in[10], (const float*)d_in[11], (const float*)d_in[13], (const float*)d_in[12], nullptr};
    GatP dec2_g{(const float*)d_in[14], (const float*)d_in[16], (const float*)d_in[15], (const float*)d_in[17], nullptr};
    FcP  enc1_f{(const float*)d_in[18], (const float*)d_in[19], (const float*)d_in[21], (const float*)d_in[20], nullptr};
    GatP enc1_g{(const float*)d_in[22], (const float*)d_in[24], (const float*)d_in[23], (const float*)d_in[25], nullptr};
    FcP  enc2_f{(const float*)d_in[26], (const float*)d_in[27], (const float*)d_in[29], (const float*)d_in[28], nullptr};
    GatP enc2_g{(const float*)d_in[30], (const float*)d_in[32], (const float*)d_in[31], (const float*)d_in[33], nullptr};
    const int* ei             = (const int*)d_in[34];
    const unsigned char* mask = (const unsigned char*)d_in[35];

    // workspace layout
    float* w = (float*)d_ws;
    size_t o = 0;
    Scratch S;
    S.hw      = w + o; o += (size_t)N_NODES * HID;
    S.gat_out = w + o; o += (size_t)N_NODES * HID;
    S.h       = w + o; o += (size_t)N_NODES * HID;
    S.zn      = w + o; o += (size_t)N_NODES * HID;
    S.emask   = w + o; o += (size_t)N_NODES * HID;
    S.scores  = w + o; o += (size_t)N_NODES * HID;
    S.alpha_s = w + o; o += N_NODES;
    S.alpha_d = w + o; o += N_NODES;
    S.mseg    = w + o; o += N_NODES;
    S.sseg    = w + o; o += N_NODES;
    S.wbuf    = w + o; o += N_EL;
    S.idx     = (int*)(w + o); o += N_NODES;
    S.mi      = (int*)(w + o); o += N_NODES;
    S.cbn     = w + o; o += 64 * 64;
    S.cbnT    = w + o; o += 64 * 64;
    S.stats   = w + o; o += 1024;
    S.acc     = w + o; o += 8;
    // f16 swizzled-weight arena (32B-aligned: offsets are multiples of 8 floats)
    _Float16* f16a = (_Float16*)(w + o);
    size_t fo = 0;
    enc1_g.Wsw = f16a + fo; fo += (size_t)DIN * HID;   // 512x64
    enc1_f.Wsw = f16a + fo; fo += (size_t)HID * HID;
    enc2_g.Wsw = f16a + fo; fo += (size_t)HID * HID;
    enc2_f.Wsw = f16a + fo; fo += (size_t)HID * HID;
    dec1_g.Wsw = f16a + fo; fo += (size_t)HID * HID;
    dec1_f.Wsw = f16a + fo; fo += (size_t)HID * HID;
    dec2_g.Wsw = f16a + fo; fo += (size_t)HID * HID;
    dec2_f.Wsw = f16a + fo; fo += (size_t)HID * DIN;   // 64x512
    S.cbnT_sw  = f16a + fo; fo += (size_t)HID * 64;
    o += (fo + 1) / 2;
    S.big     = w + o; o += (size_t)N_NODES * DIN;
    S.ei      = ei;

    float* z      = (float*)d_out;                              // [N, 64]
    float* e      = z + (size_t)N_NODES * HID;                  // [N, 64]
    float* losses = e + (size_t)N_NODES * HID;                  // 3 scalars

    // ---- one-time weight swizzles (f32 -> f16 WMMA B-fragment layout) ----
    auto swz = [&](const float* W, _Float16* dst, int k, int m) {
        int total = (k / 32) * (m / 16) * 32;
        swizzle_w_kernel<<<cdiv(total, 256), 256, 0, stream>>>(W, dst, k, m);
    };
    swz(enc1_g.W, enc1_g.Wsw, DIN, HID);
    swz(enc1_f.W, enc1_f.Wsw, HID, HID);
    swz(enc2_g.W, enc2_g.Wsw, HID, HID);
    swz(enc2_f.W, enc2_f.Wsw, HID, HID);
    swz(dec1_g.W, dec1_g.Wsw, HID, HID);
    swz(dec1_f.W, dec1_f.Wsw, HID, HID);
    swz(dec2_g.W, dec2_g.Wsw, HID, HID);
    swz(dec2_f.W, dec2_f.Wsw, HID, DIN);
    // normalized codebook (+ transpose) then swizzle for the score GEMM
    cbn_kernel<<<1, 64, 0, stream>>>(codebook, S.cbn, S.cbnT);
    swz(S.cbnT, S.cbnT_sw, HID, 64);

    // ---- encoder ----
    run_block(x,   DIN, HID, enc1_g, enc1_f, S.h, S, stream);
    run_block(S.h, HID, HID, enc2_g, enc2_f, z,   S, stream);

    // ---- vector quantizer (cosine) ----
    l2n64_kernel<<<cdiv(N_NODES, 256), 256, 0, stream>>>(z, S.zn, N_NODES);
    {
        int waves = (N_NODES / 16) * (64 / 64);
        gemm_wmma4_kernel<<<(waves + 7) / 8, 256, 0, stream>>>(S.zn, S.cbnT_sw, nullptr,
                                                               S.scores, N_NODES, HID, 64);
    }
    argmax_kernel<<<cdiv(N_NODES, 256), 256, 0, stream>>>(S.scores, S.idx, N_NODES);
    fill_f32_kernel<<<1, 8, 0, stream>>>(S.acc, 0.0f, 8);
    vq_kernel<<<cdiv(N_NODES, 256), 256, 0, stream>>>(S.zn, S.cbn, S.idx, mask,
                                                      e, S.emask, S.mi, S.acc, N_NODES);

    // ---- decode(e) + recon loss ----
    run_block(e,   HID, HID, dec1_g, dec1_f, S.h,   S, stream);
    run_block(S.h, HID, DIN, dec2_g, dec2_f, S.big, S, stream);
    mse_kernel<<<4096, 256, 0, stream>>>(S.big, x, S.acc, (long long)N_NODES * DIN);

    // ---- decode(masked e) + masked cosine loss ----
    run_block(S.emask, HID, HID, dec1_g, dec1_f, S.h,   S, stream);
    run_block(S.h,     HID, DIN, dec2_g, dec2_f, S.big, S, stream);
    maskloss_kernel<<<cdiv((long long)N_NODES * 32, 256), 256, 0, stream>>>(
        S.big, x, S.mi, S.acc, N_NODES);

    finalize_kernel<<<1, 1, 0, stream>>>(S.acc, losses);
}
